// MDAttention_T_75024488727111
// MI455X (gfx1250) — matrix operand order repaired and checked by reading, not compiled, not run on hardware
//
#include <hip/hip_runtime.h>
#include <hip/hip_bf16.h>

#define T_C   128
#define T_N   65536      // 256*256
#define T_B   4
#define HEADS 8

typedef __attribute__((ext_vector_type(16))) __bf16 v16bf;
typedef __attribute__((ext_vector_type(8)))  float  v8f;
typedef __attribute__((ext_vector_type(2)))  float  v2f;

// ---------- helpers ----------
static __device__ __forceinline__ unsigned f2bf(float f) {
  unsigned u = __float_as_uint(f);
  return (u + 0x7FFFu + ((u >> 16) & 1u)) >> 16;   // RNE f32->bf16
}
static __device__ __forceinline__ unsigned pack2bf(float lo, float hi) {
  return (f2bf(lo) & 0xFFFFu) | (f2bf(hi) << 16);
}
static __device__ __forceinline__ float gelu_f(float x) {
  return 0.5f * x * (1.0f + erff(x * 0.70710678118654752f));
}

// ---------- pack kernels ----------
// weights: consecutive pairs along K (row-major [O][C] -> [O][C/2] u32)
__global__ void k_pack_rows(const float* __restrict__ in, unsigned* __restrict__ out, int npairs) {
  int i = blockIdx.x * blockDim.x + threadIdx.x;
  if (i < npairs) out[i] = pack2bf(in[2 * i], in[2 * i + 1]);
}
// activations: pair across channel stride N ([C][N] f32 -> [C/2][N] u32)
__global__ void k_pack_chan(const float* __restrict__ in, unsigned* __restrict__ out, int total) {
  int idx = blockIdx.x * blockDim.x + threadIdx.x;
  if (idx >= total) return;
  int c2 = idx >> 16, n = idx & 65535;
  out[idx] = pack2bf(in[(size_t)(2 * c2) * T_N + n], in[(size_t)(2 * c2 + 1) * T_N + n]);
}
__global__ void k_zero(float* p, int n) {
  int i = blockIdx.x * blockDim.x + threadIdx.x;
  if (i < n) p[i] = 0.f;
}

// ---------- WMMA bf16 GEMM: out[O][N] = W[O][128] x B[128][N] ----------
// Wpk: bf16 pairs row-major [O][64] u32 ; Bpk: [64][N] u32 (pairs along C)
// EPI: 0 = f32 out, 1 = gelu + bf16-pack out, 2 = bf16-pack out, 3 = f32 out + residual
template <int EPI>
__global__ __launch_bounds__(256) void k_gemm_wmma(
    const unsigned* __restrict__ Wpk, const unsigned* __restrict__ Bpk,
    float* __restrict__ Cf, unsigned* __restrict__ Cpk,
    const float* __restrict__ Res, int N) {
  const int lane = threadIdx.x & 31;
  const int wv   = threadIdx.x >> 5;
  const int half = lane >> 4;
  const int ln   = lane & 15;
  const int m0   = blockIdx.y << 4;
  const int n0w  = (blockIdx.x * 8 + wv) * 64;

  union AB { unsigned u[8]; v16bf v; };
  AB afr[4];
  const uint4* W4 = reinterpret_cast<const uint4*>(Wpk);
  const int mrow = m0 + ln;
#pragma unroll
  for (int ks = 0; ks < 4; ++ks) {               // A 16x32 fragment per K-step
    int base = mrow * 64 + ks * 16 + 4 * half;   // u32 index, 16B aligned
    uint4 lo = W4[base >> 2];
    uint4 hi = W4[(base >> 2) + 2];
    afr[ks].u[0] = lo.x; afr[ks].u[1] = lo.y; afr[ks].u[2] = lo.z; afr[ks].u[3] = lo.w;
    afr[ks].u[4] = hi.x; afr[ks].u[5] = hi.y; afr[ks].u[6] = hi.z; afr[ks].u[7] = hi.w;
  }
#pragma unroll
  for (int sub = 0; sub < 4; ++sub) {
    const int n = n0w + sub * 16 + ln;
    v8f acc = {0.f, 0.f, 0.f, 0.f, 0.f, 0.f, 0.f, 0.f};
#pragma unroll
    for (int ks = 0; ks < 4; ++ks) {
      AB bfr;
      const int crow = ks * 16 + 4 * half;       // pair-row into [64][N]
#pragma unroll
      for (int j = 0; j < 8; ++j) bfr.u[j] = Bpk[(crow + j + (j & 4)) * N + n];
      if (ks < 3) __builtin_prefetch(&Bpk[(crow + 16) * N + n], 0, 1);
      acc = __builtin_amdgcn_wmma_f32_16x16x32_bf16(
          false, afr[ks].v, false, bfr.v, (short)0, acc, false, false);
    }
    if (EPI == 0 || EPI == 3) {
#pragma unroll
      for (int r = 0; r < 8; ++r) {
        int row = m0 + 8 * half + r;
        float v = acc[r];
        if (EPI == 3) v += Res[(size_t)row * N + n];
        Cf[(size_t)row * N + n] = v;
      }
    } else {
#pragma unroll
      for (int r = 0; r < 8; r += 2) {
        float v0 = acc[r], v1 = acc[r + 1];
        if (EPI == 1) { v0 = gelu_f(v0); v1 = gelu_f(v1); }
        int prow = (m0 >> 1) + 4 * half + (r >> 1);
        Cpk[(size_t)prow * N + n] = pack2bf(v0, v1);
      }
    }
  }
}

// ---------- depthwise 3x3 SAME ----------
__global__ void k_dwconv3(const float* __restrict__ in, const float* __restrict__ w9,
                          float* __restrict__ out, int O) {
  int idx = blockIdx.x * blockDim.x + threadIdx.x;
  if (idx >= O * T_N) return;
  int o = idx >> 16, rem = idx & 65535, y = rem >> 8, x = rem & 255;
  const float* ip = in + (size_t)o * T_N;
  const float* wp = w9 + o * 9;
  float s = 0.f;
#pragma unroll
  for (int dy = -1; dy <= 1; ++dy) {
    int yy = y + dy; if (yy < 0 || yy > 255) continue;
#pragma unroll
    for (int dx = -1; dx <= 1; ++dx) {
      int xx = x + dx; if (xx < 0 || xx > 255) continue;
      s += ip[yy * 256 + xx] * wp[(dy + 1) * 3 + (dx + 1)];
    }
  }
  out[idx] = s;
}

// ---------- l2 row norms (rows 0..255 = q then k, each [128][N]) ----------
__global__ void k_rownorm(const float* __restrict__ q, float* __restrict__ norms) {
  __shared__ float sd[256];
  const float* p = q + (size_t)blockIdx.x * T_N;
  float s = 0.f;
  for (int i = threadIdx.x; i < T_N; i += 256) { float v = p[i]; s += v * v; }
  sd[threadIdx.x] = s; __syncthreads();
  for (int st = 128; st; st >>= 1) {
    if (threadIdx.x < st) sd[threadIdx.x] += sd[threadIdx.x + st];
    __syncthreads();
  }
  if (!threadIdx.x) norms[blockIdx.x] = sqrtf(sd[0]);
}

// ---------- per-head Gram q.k^T via V_WMMA_F32_16X16X4_F32, K split ----------
#define KSLABS 64
__global__ __launch_bounds__(32) void k_gram(const float* __restrict__ qb,
                                             const float* __restrict__ kb,
                                             float* __restrict__ gram) {
  const int h = blockIdx.y, lane = threadIdx.x & 31;
  const int half = lane >> 4, ln = lane & 15;
  const float* qr = qb + (size_t)(h * 16 + ln) * T_N;
  const float* kr = kb + (size_t)(h * 16 + ln) * T_N;
  const int k0 = blockIdx.x * (T_N / KSLABS);
  v8f acc = {0.f, 0.f, 0.f, 0.f, 0.f, 0.f, 0.f, 0.f};
#if __has_builtin(__builtin_amdgcn_wmma_f32_16x16x4_f32)
  for (int k = k0; k < k0 + T_N / KSLABS; k += 4) {
    float2 aq = *reinterpret_cast<const float2*>(qr + k + 2 * half);
    float2 ak = *reinterpret_cast<const float2*>(kr + k + 2 * half);
    v2f A; A[0] = aq.x; A[1] = aq.y;
    v2f B; B[0] = ak.x; B[1] = ak.y;
    acc = __builtin_amdgcn_wmma_f32_16x16x4_f32(false, A, false, B, (short)0, acc, false, false);
  }
#else
  for (int k = k0; k < k0 + T_N / KSLABS; ++k) {
    float kv = kr[k];
#pragma unroll
    for (int r = 0; r < 8; ++r)
      acc[r] += qb[(size_t)(h * 16 + 8 * half + r) * T_N + k] * kv;
  }
#endif
#pragma unroll
  for (int r = 0; r < 8; ++r)
    atomicAdd(&gram[h * 256 + (8 * half + r) * 16 + ln], acc[r]);
}

__global__ void k_softmax(const float* __restrict__ gram, const float* __restrict__ norms,
                          const float* __restrict__ temp, float* __restrict__ attn) {
  __shared__ float L[256], E[256];
  int h = blockIdx.x, t = threadIdx.x, c = t >> 4, d = t & 15;
  float nq = fmaxf(norms[h * 16 + c], 1e-12f);
  float nk = fmaxf(norms[128 + h * 16 + d], 1e-12f);
  float lg = gram[h * 256 + t] / (nq * nk) * temp[h];
  L[t] = lg; __syncthreads();
  float mx = -3.0e38f;
#pragma unroll
  for (int j = 0; j < 16; ++j) mx = fmaxf(mx, L[c * 16 + j]);
  float e = expf(lg - mx);
  E[t] = e; __syncthreads();
  float sm = 0.f;
#pragma unroll
  for (int j = 0; j < 16; ++j) sm += E[c * 16 + j];
  attn[h * 256 + t] = e / sm;
}

__global__ void k_attnout(const float* __restrict__ attn, const float* __restrict__ v,
                          float* __restrict__ out) {
  __shared__ float A[256];
  int h = blockIdx.y, n = blockIdx.x * 256 + threadIdx.x;
  A[threadIdx.x] = attn[h * 256 + threadIdx.x]; __syncthreads();
  float vd[16];
#pragma unroll
  for (int d = 0; d < 16; ++d) vd[d] = v[(size_t)(h * 16 + d) * T_N + n];
#pragma unroll
  for (int c = 0; c < 16; ++c) {
    float s = 0.f;
#pragma unroll
    for (int d = 0; d < 16; ++d) s += A[c * 16 + d] * vd[d];
    out[(size_t)(h * 16 + c) * T_N + n] = s;
  }
}

// ---------- 256-pt radix-2 FFT (DIT, LDS), 64 threads ----------
static __device__ void fft256(float* re, float* im, int t) {
  for (int s = 1; s <= 8; ++s) {
    int m = 1 << s, mh = m >> 1;
    for (int b = t; b < 128; b += 64) {
      int pos = b & (mh - 1);
      int i0 = ((b >> (s - 1)) << s) + pos;
      int i1 = i0 + mh;
      float ang = -6.283185307179586f * (float)pos / (float)m;
      float wr = cosf(ang), wi = sinf(ang);
      float xr = re[i1], xi = im[i1];
      float tr = wr * xr - wi * xi, ti = wr * xi + wi * xr;
      re[i1] = re[i0] - tr; im[i1] = im[i0] - ti;
      re[i0] += tr;         im[i0] += ti;
    }
    __syncthreads();
  }
}
__global__ __launch_bounds__(64) void k_fft_rows_r2c(const float* __restrict__ in,
                                                     float2* __restrict__ out) {
  __shared__ float re[256], im[256];
  const float* ip = in + (size_t)blockIdx.x * 256;
  float2* op = out + (size_t)blockIdx.x * 256;
  int t = threadIdx.x;
  for (int i = t; i < 256; i += 64) { int r = __brev((unsigned)i) >> 24; re[r] = ip[i]; im[r] = 0.f; }
  __syncthreads();
  fft256(re, im, t);
  for (int i = t; i < 256; i += 64) op[i] = make_float2(re[i], im[i]);
}
__global__ __launch_bounds__(64) void k_fft_cols_c2r(const float2* __restrict__ in,
                                                     float* __restrict__ out, float scale) {
  __shared__ float re[256], im[256];
  int c = blockIdx.x >> 8, x = blockIdx.x & 255, t = threadIdx.x;
  const float2* ip = in + (size_t)c * T_N + x;
  float* op = out + (size_t)c * T_N + x;
  for (int i = t; i < 256; i += 64) {
    int r = __brev((unsigned)i) >> 24;
    float2 v = ip[(size_t)i * 256];
    re[r] = v.x; im[r] = v.y;
  }
  __syncthreads();
  fft256(re, im, t);
  for (int i = t; i < 256; i += 64) op[(size_t)i * 256] = re[i] * scale;
}

// ---------- 8x8 per-patch circular convolution (== rfft2*rfft2 -> irfft2) ----------
__global__ void k_circconv(const float* __restrict__ vf, const float* __restrict__ kf,
                           float* __restrict__ out) {
  __shared__ float sv[4][64], sk[4][64];
  int lp = threadIdx.x >> 6, p = threadIdx.x & 63, i = p >> 3, j = p & 7;
  int patch = blockIdx.x * 4 + lp;
  int c = patch >> 10, g = patch & 1023, gy = g >> 5, gx = g & 31;
  size_t base = (size_t)c * T_N + (size_t)gy * 8 * 256 + gx * 8;
  sv[lp][p] = vf[base + (size_t)i * 256 + j];
  sk[lp][p] = kf[base + (size_t)i * 256 + j];
  __syncthreads();
  float s = 0.f;
#pragma unroll
  for (int u = 0; u < 8; ++u)
#pragma unroll
    for (int v = 0; v < 8; ++v)
      s += sv[lp][u * 8 + v] * sk[lp][((i - u) & 7) * 8 + ((j - v) & 7)];
  out[base + (size_t)i * 256 + j] = s;
}

// ---------- channel LayerNorm * qf, emit packed bf16 for final GEMM ----------
__global__ void k_ln_mul_pack(const float* __restrict__ x, const float* __restrict__ qf,
                              const float* __restrict__ w, const float* __restrict__ b,
                              unsigned* __restrict__ outpk) {
  int n = blockIdx.x * 256 + threadIdx.x;
  float mu = 0.f;
  for (int c = 0; c < 128; ++c) mu += x[(size_t)c * T_N + n];
  mu *= (1.f / 128.f);
  float var = 0.f;
  for (int c = 0; c < 128; ++c) { float d = x[(size_t)c * T_N + n] - mu; var += d * d; }
  var *= (1.f / 128.f);
  float rs = rsqrtf(var + 1e-5f);
  for (int c2 = 0; c2 < 64; ++c2) {
    int c0 = 2 * c2;
    float v0 = ((x[(size_t)c0 * T_N + n] - mu) * rs * w[c0] + b[c0]) * qf[(size_t)c0 * T_N + n];
    float v1 = ((x[(size_t)(c0 + 1) * T_N + n] - mu) * rs * w[c0 + 1] + b[c0 + 1]) *
               qf[(size_t)(c0 + 1) * T_N + n];
    outpk[(size_t)c2 * T_N + n] = pack2bf(v0, v1);
  }
}

// =======================================================================
extern "C" void kernel_launch(void* const* d_in, const int* in_sizes, int n_in,
                              void* d_out, int out_size, void* d_ws, size_t ws_size,
                              hipStream_t stream) {
  (void)in_sizes; (void)n_in; (void)out_size; (void)ws_size;
  const float* x       = (const float*)d_in[0];
  const float* qkv_w   = (const float*)d_in[1];
  const float* qkv_dw  = (const float*)d_in[2];
  const float* proj_w  = (const float*)d_in[3];
  const float* q1_w    = (const float*)d_in[4];
  const float* q2_w    = (const float*)d_in[5];
  const float* kv_w    = (const float*)d_in[6];
  const float* kv_dw   = (const float*)d_in[7];
  const float* projf_w = (const float*)d_in[8];
  const float* temp    = (const float*)d_in[9];
  const float* ln_w    = (const float*)d_in[10];
  const float* ln_b    = (const float*)d_in[11];
  float* out = (float*)d_out;

  const size_t t = (size_t)T_C * T_N;             // 8,388,608 floats / batch
  float* F  = (float*)d_ws;
  float* s0 = F;                                  // 3t (qkv raw / FFT complex+real)
  float* s1 = F + 3 * t;                          // 3t (qkv dw / kv raw / out_1)
  float* s2 = F + 6 * t;                          // t  (attnout / x_fft3 / vf)
  float* s3 = F + 7 * t;                          // t  (channel-attn out, residual)
  unsigned* P  = (unsigned*)(F + 8 * t);
  unsigned* p0 = P;                               // t/2 u32 each
  unsigned* p1 = P + t / 2;
  unsigned* p2 = P + t;
  unsigned* wpk     = P + 3 * (t / 2);
  unsigned* w_qkv   = wpk;                        // 384*64
  unsigned* w_proj  = wpk + 24576;                // 128*64
  unsigned* w_q1    = wpk + 32768;
  unsigned* w_q2    = wpk + 40960;
  unsigned* w_kv    = wpk + 49152;                // 256*64
  unsigned* w_projf = wpk + 65536;
  float* smallf = (float*)(wpk + 73728);
  float* norms = smallf;                          // 256
  float* gram  = smallf + 256;                    // 2048
  float* attn  = smallf + 2304;                   // 2048

  // weights -> bf16 pairs (row-major): done once per call, tiny
  k_pack_rows<<<(24576 + 255) / 256, 256, 0, stream>>>(qkv_w,   w_qkv,   24576);
  k_pack_rows<<<(8192  + 255) / 256, 256, 0, stream>>>(proj_w,  w_proj,  8192);
  k_pack_rows<<<(8192  + 255) / 256, 256, 0, stream>>>(q1_w,    w_q1,    8192);
  k_pack_rows<<<(8192  + 255) / 256, 256, 0, stream>>>(q2_w,    w_q2,    8192);
  k_pack_rows<<<(16384 + 255) / 256, 256, 0, stream>>>(kv_w,    w_kv,    16384);
  k_pack_rows<<<(8192  + 255) / 256, 256, 0, stream>>>(projf_w, w_projf, 8192);

  const int PKT = 64 * T_N;                       // packed elements per activation
  for (int b = 0; b < T_B; ++b) {
    const float* xb  = x  + (size_t)b * t;
    float*       ob  = out + (size_t)b * t;
    // ---- channel attention ----
    k_pack_chan<<<16384, 256, 0, stream>>>(xb, p0, PKT);
    k_gemm_wmma<0><<<dim3(128, 24), 256, 0, stream>>>(w_qkv, p0, s0, nullptr, nullptr, T_N); // qkv
    k_dwconv3<<<(384 * T_N + 255) / 256, 256, 0, stream>>>(s0, qkv_dw, s1, 384);
    k_rownorm<<<256, 256, 0, stream>>>(s1, norms);                       // q rows 0-127, k 128-255
    k_zero<<<8, 256, 0, stream>>>(gram, 2048);
    k_gram<<<dim3(KSLABS, HEADS), 32, 0, stream>>>(s1, s1 + t, gram);    // f32 WMMA Gram
    k_softmax<<<HEADS, 256, 0, stream>>>(gram, norms, temp, attn);
    k_attnout<<<dim3(256, HEADS), 256, 0, stream>>>(attn, s1 + 2 * t, s2);
    k_pack_chan<<<16384, 256, 0, stream>>>(s2, p0, PKT);
    k_gemm_wmma<0><<<dim3(128, 8), 256, 0, stream>>>(w_proj, p0, s3, nullptr, nullptr, T_N); // out
    k_pack_chan<<<16384, 256, 0, stream>>>(s3, p1, PKT);                 // out (bf16) for kv GEMM
    // ---- FFT gating branch ----
    k_fft_rows_r2c<<<32768, 64, 0, stream>>>(xb, (float2*)s0);
    k_fft_cols_c2r<<<32768, 64, 0, stream>>>((const float2*)s0, s0 + 2 * t, 1.0f);  // R = Re(FFT2 x)
    k_pack_chan<<<16384, 256, 0, stream>>>(s0 + 2 * t, p0, PKT);
    k_gemm_wmma<1><<<dim3(128, 8), 256, 0, stream>>>(w_q1, p0, nullptr, p2, nullptr, T_N); // gelu+pack
    k_gemm_wmma<0><<<dim3(128, 8), 256, 0, stream>>>(w_q2, p2, s2, nullptr, nullptr, T_N); // x_fft3
    k_fft_rows_r2c<<<32768, 64, 0, stream>>>(s2, (float2*)s0);
    k_fft_cols_c2r<<<32768, 64, 0, stream>>>((const float2*)s0, s0 + 2 * t,
                                             1.0f / 65536.0f);           // qf = Re(ifft2)
    k_gemm_wmma<0><<<dim3(128, 16), 256, 0, stream>>>(w_kv, p1, s1, nullptr, nullptr, T_N); // kv raw
    k_dwconv3<<<(128 * T_N + 255) / 256, 256, 0, stream>>>(s1,     kv_dw,           s1 + 2 * t, 128); // kf
    k_dwconv3<<<(128 * T_N + 255) / 256, 256, 0, stream>>>(s1 + t, kv_dw + 128 * 9, s2,         128); // vf
    k_circconv<<<32768, 256, 0, stream>>>(s2, s1 + 2 * t, s1);           // out_1
    k_ln_mul_pack<<<256, 256, 0, stream>>>(s1, s0 + 2 * t, ln_w, ln_b, p0);
    k_gemm_wmma<3><<<dim3(128, 8), 256, 0, stream>>>(w_projf, p0, ob, nullptr, s3, T_N); // + residual
  }
}
